// E_GCL_43851616092222
// MI455X (gfx1250) — matrix operand order repaired and checked by reading, not compiled, not run on hardware
//
#include <hip/hip_runtime.h>

#define N_NODES 10000
#define N_EDGES 320000
#define EPSF    1e-8f
#define LN_EPSF 1e-5f

typedef __attribute__((ext_vector_type(16))) __bf16 v16bf;
typedef __attribute__((ext_vector_type(8)))  float  v8f;
typedef unsigned int u32x4 __attribute__((ext_vector_type(4)));
typedef int          i32x8 __attribute__((ext_vector_type(8)));
typedef int          i32x4 __attribute__((ext_vector_type(4)));

union BFV { uint4 u[2]; v16bf v; };
union US8 { unsigned short s[8]; uint4 u; };

static __device__ __forceinline__ unsigned short f2bf(float f) {
  unsigned int u = __float_as_uint(f);
  return (unsigned short)((u + 0x7FFFu + ((u >> 16) & 1u)) >> 16);
}

static __device__ __forceinline__ v8f zero8() {
  v8f z = {0.f, 0.f, 0.f, 0.f, 0.f, 0.f, 0.f, 0.f};
  return z;
}

static __device__ __forceinline__ float silu_f(float x) {
  return x / (1.f + __expf(-x));
}

// A operand: 16x32 bf16 tile from row-major LDS, per ISA 7.12.2:
// lanes 0-15 (kh=0): elems 0..7 = K+0..7,  8..15 = K+16..23
// lanes 16-31 (kh=1): elems 0..7 = K+8..15, 8..15 = K+24..31 ; M = lane&15
static __device__ __forceinline__ v16bf load_a(const unsigned short* rowp, int kofs, int kh) {
  BFV t;
  const unsigned short* p = rowp + kofs + kh * 8;
  t.u[0] = *(const uint4*)(p);
  t.u[1] = *(const uint4*)(p + 16);
  return t.v;
}

// B operand: 32x16 bf16 tile from [32][256] LDS panel: N = lane&15, elems 0..7 = K+8*kh+0..7, 8..15 = K+16+8*kh+0..7.
static __device__ __forceinline__ v16bf load_b(const unsigned short* Bst, int lane, int col0) {
  union { unsigned short s[16]; v16bf v; } t;
  const unsigned short* p = Bst + (lane >> 4) * 8 * 256 + col0 + (lane & 15);
#pragma unroll
  for (int i = 0; i < 8; ++i) {
    t.s[i] = p[i * 256];
    t.s[8 + i] = p[(16 + i) * 256];
  }
  return t.v;
}

static __device__ __forceinline__ v8f wmma_bf16(v16bf a, v16bf b, v8f c) {
  return __builtin_amdgcn_wmma_f32_16x16x32_bf16(false, a, false, b, (short)0, c, false, false);
}

// ---- Tensor Data Mover: DMA one [32][256] bf16 weight panel into LDS ----
// D# per CDNA5 ISA ch.8: group0 = {count/type | lds_addr | global_addr},
// group1 = {data_size, tensor_dim0/1, tile_dim0/1, tensor_dim0_stride}, 2D so
// remaining groups are zero. Tracked by TENSORcnt; one wave issues, EXEC ignored.
// This toolchain's builtin takes 6 args (g0, g1, g2, g3, g4, cpol).
static __device__ __forceinline__ void tdm_issue_b(unsigned ldsOff,
                                                   const unsigned short* Wg, int kk) {
  unsigned long long ga =
      (unsigned long long)(uintptr_t)(Wg + (size_t)kk * (32 * 256));
  u32x4 g0;
  g0[0] = 1u;                                                    // count=1 (valid user D#)
  g0[1] = ldsOff;                                                // lds_addr (bytes)
  g0[2] = (unsigned)(ga & 0xFFFFFFFFull);                        // global_addr[31:0]
  g0[3] = (unsigned)((ga >> 32) & 0x01FFFFFFull) | (2u << 30);   // addr[56:32] | type=2
  i32x8 g1;
  g1[0] = 0x00010000;          // workgroup_mask=0, data_size=1 (2B), no pad/iterate
  g1[1] = (int)(256u << 16);   // tensor_dim0 = 256 (bits 79:48, low half)
  g1[2] = (int)(32u << 16);    // tensor_dim0 hi | tensor_dim1 = 32 (low half)
  g1[3] = (int)(256u << 16);   // tensor_dim1 hi | tile_dim0 = 256
  g1[4] = 32;                  // tile_dim1 = 32, tile_dim2 = 0
  g1[5] = 256;                 // tensor_dim0_stride = 256 elements
  g1[6] = 0;
  g1[7] = 0;
  i32x4 gz4 = {0, 0, 0, 0};
  i32x8 gz8 = {0, 0, 0, 0, 0, 0, 0, 0};
  __builtin_amdgcn_tensor_load_to_lds(g0, g1, gz4, gz4, gz8, 0);
}

// ---------------- prep kernels ----------------

__global__ void zero_f32_kernel(float* __restrict__ p, int n) {
  int i = blockIdx.x * blockDim.x + threadIdx.x;
  if (i < n) p[i] = 0.f;
}

__global__ void cvt_pad_kernel(const float* __restrict__ src, unsigned short* __restrict__ dst,
                               int src_rows, int dst_rows, int cols) {
  long total = (long)dst_rows * cols;
  for (long i = (long)blockIdx.x * blockDim.x + threadIdx.x; i < total;
       i += (long)gridDim.x * blockDim.x) {
    int r = (int)(i / cols);
    dst[i] = (r < src_rows) ? f2bf(src[i]) : (unsigned short)0;
  }
}

// ---------------- edge kernel ----------------
// Block: 256 threads (8 waves), 64 edges. Wave w: M-tile = w>>1, N-half = w&1.
// Weight panels stream via TDM, double-buffered: DMA of panel k+1 overlaps
// WMMA sweep over panel k; wave 0 waits TENSORcnt<=1 (in-order completion).
__global__ __launch_bounds__(256) void egcl_edge_kernel(
    const float* __restrict__ coord,
    const int*   __restrict__ eidx,
    const unsigned short* __restrict__ hbf,     // [N][256] bf16
    const unsigned short* __restrict__ We1p,    // [544][256] bf16 (row 512 = radial, 513..543 zero)
    const float* __restrict__ be1,
    const unsigned short* __restrict__ We2b,    // [256][256]
    const float* __restrict__ be2,
    const unsigned short* __restrict__ Wc1b,    // [256][256]
    const float* __restrict__ bc1,
    const float* __restrict__ Wc2,              // [256] fp32
    float* __restrict__ eij_out,                // [E][256]
    float* __restrict__ agg,                    // [N][256] fp32
    float* __restrict__ agg_c,                  // [N][3]
    float* __restrict__ cnt)                    // [N]
{
  __shared__ __align__(16) unsigned short Ast[64 * 32];
  __shared__ __align__(16) unsigned short Bst0[32 * 256];
  __shared__ __align__(16) unsigned short Bst1[32 * 256];
  __shared__ __align__(16) unsigned short Hid[64 * 256];
  __shared__ float cdL[64 * 3];
  __shared__ float radL[64];
  __shared__ float dotL[64];
  __shared__ int   rowL[64];
  __shared__ int   colL[64];

  const int tid  = threadIdx.x;
  const int lane = tid & 31;
  const int wave = tid >> 5;
  const int mt   = wave >> 1;       // M-tile 0..3
  const int nh   = wave & 1;        // N-half
  const int am   = lane & 15;       // A row (M) / C column within tile
  const int kh   = lane >> 4;       // lane half
  const int e0   = blockIdx.x * 64; // N_EDGES % 64 == 0

  const unsigned bOff0 = (unsigned)(uintptr_t)(void*)Bst0;
  const unsigned bOff1 = (unsigned)(uintptr_t)(void*)Bst1;

  if (tid < 64) {
    int e = e0 + tid;
    int r = eidx[e];
    int c = eidx[N_EDGES + e];
    rowL[tid] = r;
    colL[tid] = c;
    float dx = coord[r * 3 + 0] - coord[c * 3 + 0];
    float dy = coord[r * 3 + 1] - coord[c * 3 + 1];
    float dz = coord[r * 3 + 2] - coord[c * 3 + 2];
    cdL[tid * 3 + 0] = dx;
    cdL[tid * 3 + 1] = dy;
    cdL[tid * 3 + 2] = dz;
    radL[tid] = dx * dx + dy * dy + dz * dz;
    dotL[tid] = 0.f;
  }
  __syncthreads();

  v8f acc[8];
#pragma unroll
  for (int j = 0; j < 8; ++j) acc[j] = zero8();

  const int eA = tid >> 2, qA = tid & 3;

  // ---- GEMM1: e_in[64x544] @ We1p[544x256], SiLU -> Hid ----
  if (wave == 0) tdm_issue_b(bOff0, We1p, 0);
  for (int kk = 0; kk < 17; ++kk) {
    __syncthreads();   // prior reads of the buffer being re-filled are done
    {
      int k0 = kk * 32 + qA * 8;
      uint4 val;
      if (k0 < 256) {
        val = *(const uint4*)(hbf + (size_t)rowL[eA] * 256 + k0);
      } else if (k0 < 512) {
        val = *(const uint4*)(hbf + (size_t)colL[eA] * 256 + (k0 - 256));
      } else {
        US8 t;
#pragma unroll
        for (int i = 0; i < 8; ++i) t.s[i] = 0;
        if (k0 == 512) t.s[0] = f2bf(radL[eA]);
        val = t.u;
      }
      ((uint4*)Ast)[tid] = val;
    }
    if (wave == 0) {
      if (kk + 1 < 17) {
        tdm_issue_b((kk & 1) ? bOff0 : bOff1, We1p, kk + 1);
        __builtin_amdgcn_s_wait_tensorcnt(1);   // panel kk has landed
      } else {
        __builtin_amdgcn_s_wait_tensorcnt(0);
      }
    }
    __syncthreads();
    const unsigned short* Bcur = (kk & 1) ? Bst1 : Bst0;
    v16bf a = load_a(Ast + (mt * 16 + am) * 32, 0, kh);
#pragma unroll
    for (int j = 0; j < 8; ++j) {
      v16bf b = load_b(Bcur, lane, (nh * 8 + j) * 16);
      acc[j] = wmma_bf16(a, b, acc[j]);
    }
  }
#pragma unroll
  for (int j = 0; j < 8; ++j) {
    int N = (nh * 8 + j) * 16 + am;
    float bb = be1[N];
#pragma unroll
    for (int r = 0; r < 8; ++r) {
      float s = silu_f(acc[j][r] + bb);
      Hid[(mt * 16 + r + kh * 8) * 256 + N] = f2bf(s);
    }
    acc[j] = zero8();
  }
  __syncthreads();

  // ---- GEMM2: Hid @ We2, SiLU -> e_ij (store + agg atomics + restash in Hid) ----
  if (wave == 0) tdm_issue_b(bOff0, We2b, 0);
  for (int kk = 0; kk < 8; ++kk) {
    __syncthreads();
    if (wave == 0) {
      if (kk + 1 < 8) {
        tdm_issue_b((kk & 1) ? bOff0 : bOff1, We2b, kk + 1);
        __builtin_amdgcn_s_wait_tensorcnt(1);
      } else {
        __builtin_amdgcn_s_wait_tensorcnt(0);
      }
    }
    __syncthreads();
    const unsigned short* Bcur = (kk & 1) ? Bst1 : Bst0;
    v16bf a = load_a(Hid + (mt * 16 + am) * 256, kk * 32, kh);
#pragma unroll
    for (int j = 0; j < 8; ++j) {
      v16bf b = load_b(Bcur, lane, (nh * 8 + j) * 16);
      acc[j] = wmma_bf16(a, b, acc[j]);
    }
  }
  __syncthreads();   // all Hid reads done before overwrite
#pragma unroll
  for (int j = 0; j < 8; ++j) {
    int N = (nh * 8 + j) * 16 + am;
    float bb = be2[N];
#pragma unroll
    for (int r = 0; r < 8; ++r) {
      int erow = mt * 16 + r + kh * 8;
      float s = silu_f(acc[j][r] + bb);
      eij_out[(size_t)(e0 + erow) * 256 + N] = s;
      unsafeAtomicAdd(&agg[(size_t)rowL[erow] * 256 + N], s);
      Hid[erow * 256 + N] = f2bf(s);
    }
    acc[j] = zero8();
  }
  __syncthreads();

  // ---- GEMM3: e_ij @ Wc1, SiLU -> c_hid, dot with Wc2 ----
  if (wave == 0) tdm_issue_b(bOff0, Wc1b, 0);
  for (int kk = 0; kk < 8; ++kk) {
    __syncthreads();
    if (wave == 0) {
      if (kk + 1 < 8) {
        tdm_issue_b((kk & 1) ? bOff0 : bOff1, Wc1b, kk + 1);
        __builtin_amdgcn_s_wait_tensorcnt(1);
      } else {
        __builtin_amdgcn_s_wait_tensorcnt(0);
      }
    }
    __syncthreads();
    const unsigned short* Bcur = (kk & 1) ? Bst1 : Bst0;
    v16bf a = load_a(Hid + (mt * 16 + am) * 256, kk * 32, kh);
#pragma unroll
    for (int j = 0; j < 8; ++j) {
      v16bf b = load_b(Bcur, lane, (nh * 8 + j) * 16);
      acc[j] = wmma_bf16(a, b, acc[j]);
    }
  }
  float p[8] = {0.f, 0.f, 0.f, 0.f, 0.f, 0.f, 0.f, 0.f};
#pragma unroll
  for (int j = 0; j < 8; ++j) {
    int N = (nh * 8 + j) * 16 + am;
    float bb = bc1[N];
    float w  = Wc2[N];
#pragma unroll
    for (int r = 0; r < 8; ++r) p[r] += silu_f(acc[j][r] + bb) * w;
  }
#pragma unroll
  for (int r = 0; r < 8; ++r)
    unsafeAtomicAdd(&dotL[mt * 16 + r + kh * 8], p[r]);   // ds_add_f32
  __syncthreads();

  if (tid < 64) {
    float c   = dotL[tid];
    float inv = 1.f / (sqrtf(radL[tid]) + EPSF);
    int   r   = rowL[tid];
    unsafeAtomicAdd(&agg_c[(size_t)r * 3 + 0], cdL[tid * 3 + 0] * inv * c);
    unsafeAtomicAdd(&agg_c[(size_t)r * 3 + 1], cdL[tid * 3 + 1] * inv * c);
    unsafeAtomicAdd(&agg_c[(size_t)r * 3 + 2], cdL[tid * 3 + 2] * inv * c);
    unsafeAtomicAdd(&cnt[r], 1.f);
  }
}

// ---------------- node kernel ----------------
__global__ __launch_bounds__(256) void egcl_node_kernel(
    const float* __restrict__ h,
    const float* __restrict__ coord,
    const unsigned short* __restrict__ hbf,     // [N][256] bf16
    const float* __restrict__ agg,              // [N][256] fp32
    const unsigned short* __restrict__ Wn1b,    // [512][256]
    const float* __restrict__ bn1,
    const unsigned short* __restrict__ Wn2b,    // [256][256]
    const float* __restrict__ bn2,
    const float* __restrict__ ln_g,
    const float* __restrict__ ln_b,
    const float* __restrict__ agg_c,
    const float* __restrict__ cnt,
    float* __restrict__ h_out,                  // [N][256]
    float* __restrict__ coord_out)              // [N][3]
{
  __shared__ __align__(16) unsigned short Ast[64 * 32];
  __shared__ __align__(16) unsigned short Bst0[32 * 256];
  __shared__ __align__(16) unsigned short Bst1[32 * 256];
  __shared__ __align__(16) unsigned short Hid[64 * 256];
  __shared__ float sumL[64];
  __shared__ float sqL[64];

  const int tid  = threadIdx.x;
  const int lane = tid & 31;
  const int wave = tid >> 5;
  const int mt   = wave >> 1;
  const int nh   = wave & 1;
  const int am   = lane & 15;
  const int kh   = lane >> 4;
  const int n0   = blockIdx.x * 64;

  const unsigned bOff0 = (unsigned)(uintptr_t)(void*)Bst0;
  const unsigned bOff1 = (unsigned)(uintptr_t)(void*)Bst1;

  if (tid < 64) { sumL[tid] = 0.f; sqL[tid] = 0.f; }

  v8f acc[8];
#pragma unroll
  for (int j = 0; j < 8; ++j) acc[j] = zero8();

  const int eA = tid >> 2, qA = tid & 3;

  // ---- GEMM1: [h|agg][64x512] @ Wn1[512x256], SiLU -> Hid ----
  if (wave == 0) tdm_issue_b(bOff0, Wn1b, 0);
  for (int kk = 0; kk < 16; ++kk) {
    __syncthreads();
    {
      int node = n0 + eA;
      int k0 = kk * 32 + qA * 8;
      uint4 val;
      if (node >= N_NODES) {
        US8 t;
#pragma unroll
        for (int i = 0; i < 8; ++i) t.s[i] = 0;
        val = t.u;
      } else if (k0 < 256) {
        val = *(const uint4*)(hbf + (size_t)node * 256 + k0);
      } else {
        const float* s = agg + (size_t)node * 256 + (k0 - 256);
        US8 t;
#pragma unroll
        for (int i = 0; i < 8; ++i) t.s[i] = f2bf(s[i]);
        val = t.u;
      }
      ((uint4*)Ast)[tid] = val;
    }
    if (wave == 0) {
      if (kk + 1 < 16) {
        tdm_issue_b((kk & 1) ? bOff0 : bOff1, Wn1b, kk + 1);
        __builtin_amdgcn_s_wait_tensorcnt(1);
      } else {
        __builtin_amdgcn_s_wait_tensorcnt(0);
      }
    }
    __syncthreads();
    const unsigned short* Bcur = (kk & 1) ? Bst1 : Bst0;
    v16bf a = load_a(Ast + (mt * 16 + am) * 32, 0, kh);
#pragma unroll
    for (int j = 0; j < 8; ++j) {
      v16bf b = load_b(Bcur, lane, (nh * 8 + j) * 16);
      acc[j] = wmma_bf16(a, b, acc[j]);
    }
  }
#pragma unroll
  for (int j = 0; j < 8; ++j) {
    int N = (nh * 8 + j) * 16 + am;
    float bb = bn1[N];
#pragma unroll
    for (int r = 0; r < 8; ++r) {
      float s = silu_f(acc[j][r] + bb);
      Hid[(mt * 16 + r + kh * 8) * 256 + N] = f2bf(s);
    }
    acc[j] = zero8();
  }
  __syncthreads();

  // ---- GEMM2: Hid @ Wn2 + bn2 + h (residual), then LayerNorm ----
  if (wave == 0) tdm_issue_b(bOff0, Wn2b, 0);
  for (int kk = 0; kk < 8; ++kk) {
    __syncthreads();
    if (wave == 0) {
      if (kk + 1 < 8) {
        tdm_issue_b((kk & 1) ? bOff0 : bOff1, Wn2b, kk + 1);
        __builtin_amdgcn_s_wait_tensorcnt(1);
      } else {
        __builtin_amdgcn_s_wait_tensorcnt(0);
      }
    }
    __syncthreads();
    const unsigned short* Bcur = (kk & 1) ? Bst1 : Bst0;
    v16bf a = load_a(Hid + (mt * 16 + am) * 256, kk * 32, kh);
#pragma unroll
    for (int j = 0; j < 8; ++j) {
      v16bf b = load_b(Bcur, lane, (nh * 8 + j) * 16);
      acc[j] = wmma_bf16(a, b, acc[j]);
    }
  }
  float s1[8] = {0.f, 0.f, 0.f, 0.f, 0.f, 0.f, 0.f, 0.f};
  float s2[8] = {0.f, 0.f, 0.f, 0.f, 0.f, 0.f, 0.f, 0.f};
#pragma unroll
  for (int j = 0; j < 8; ++j) {
    int N = (nh * 8 + j) * 16 + am;
    float b2 = bn2[N];
#pragma unroll
    for (int r = 0; r < 8; ++r) {
      int row  = mt * 16 + r + kh * 8;
      int node = n0 + row;
      float v = acc[j][r] + b2;
      if (node < N_NODES) v += h[(size_t)node * 256 + N];
      acc[j][r] = v;
      s1[r] += v;
      s2[r] += v * v;
    }
  }
#pragma unroll
  for (int r = 0; r < 8; ++r) {
    int row = mt * 16 + r + kh * 8;
    unsafeAtomicAdd(&sumL[row], s1[r]);   // ds_add_f32
    unsafeAtomicAdd(&sqL[row], s2[r]);
  }
  __syncthreads();

  float mu[8], rs[8];
#pragma unroll
  for (int r = 0; r < 8; ++r) {
    int row = mt * 16 + r + kh * 8;
    float m   = sumL[row] * (1.f / 256.f);
    float var = sqL[row] * (1.f / 256.f) - m * m;
    mu[r] = m;
    rs[r] = rsqrtf(var + LN_EPSF);
  }
#pragma unroll
  for (int j = 0; j < 8; ++j) {
    int N = (nh * 8 + j) * 16 + am;
    float g  = ln_g[N];
    float bb = ln_b[N];
#pragma unroll
    for (int r = 0; r < 8; ++r) {
      int node = n0 + mt * 16 + r + kh * 8;
      if (node < N_NODES)
        h_out[(size_t)node * 256 + N] = (acc[j][r] - mu[r]) * rs[r] * g + bb;
    }
  }

  if (tid < 64) {
    int node = n0 + tid;
    if (node < N_NODES) {
      float inv = 1.f / (cnt[node] + EPSF);
#pragma unroll
      for (int c = 0; c < 3; ++c)
        coord_out[(size_t)node * 3 + c] =
            coord[(size_t)node * 3 + c] + agg_c[(size_t)node * 3 + c] * inv;
    }
  }
}

// ---------------- launch ----------------

extern "C" void kernel_launch(void* const* d_in, const int* in_sizes, int n_in,
                              void* d_out, int out_size, void* d_ws, size_t ws_size,
                              hipStream_t stream) {
  (void)in_sizes; (void)n_in; (void)out_size; (void)ws_size;

  const float* h     = (const float*)d_in[0];
  const float* coord = (const float*)d_in[1];
  const int*   eidx  = (const int*)d_in[2];
  const float* We1   = (const float*)d_in[3];
  const float* be1   = (const float*)d_in[4];
  const float* We2   = (const float*)d_in[5];
  const float* be2   = (const float*)d_in[6];
  const float* Wn1   = (const float*)d_in[7];
  const float* bn1   = (const float*)d_in[8];
  const float* Wn2   = (const float*)d_in[9];
  const float* bn2   = (const float*)d_in[10];
  const float* ln_g  = (const float*)d_in[11];
  const float* ln_b  = (const float*)d_in[12];
  const float* Wc1   = (const float*)d_in[13];
  const float* bc1   = (const float*)d_in[14];
  const float* Wc2   = (const float*)d_in[15];

  float* out       = (float*)d_out;
  float* h_out     = out;                                   // [10000][256]
  float* coord_out = out + (size_t)N_NODES * 256;           // [10000][3]
  float* eij_out   = coord_out + (size_t)N_NODES * 3;       // [320000][256]

  // workspace layout (16B aligned offsets), ~16.5 MB total
  char* ws = (char*)d_ws;
  float* agg   = (float*)(ws + 0);            // 10,240,000 B
  float* agg_c = (float*)(ws + 10240000);     //    120,000 B
  float* cnt   = (float*)(ws + 10360000);     //     40,000 B
  unsigned short* hbf  = (unsigned short*)(ws + 10400000);  // 5,120,000 B
  unsigned short* We1p = (unsigned short*)(ws + 15520000);  //   278,528 B
  unsigned short* We2b = (unsigned short*)(ws + 15798528);  //   131,072 B
  unsigned short* Wc1b = (unsigned short*)(ws + 15929600);  //   131,072 B
  unsigned short* Wn1b = (unsigned short*)(ws + 16060672);  //   262,144 B
  unsigned short* Wn2b = (unsigned short*)(ws + 16322816);  //   131,072 B

  // zero accumulators (agg | agg_c | cnt contiguous = 2,600,000 floats)
  zero_f32_kernel<<<(2600000 + 255) / 256, 256, 0, stream>>>(agg, 2600000);

  // bf16 conversions (weights stay L2-resident afterwards)
  cvt_pad_kernel<<<10000, 256, 0, stream>>>(h,   hbf,  N_NODES, N_NODES, 256);
  cvt_pad_kernel<<<544,   256, 0, stream>>>(We1, We1p, 513, 544, 256);
  cvt_pad_kernel<<<256,   256, 0, stream>>>(We2, We2b, 256, 256, 256);
  cvt_pad_kernel<<<256,   256, 0, stream>>>(Wc1, Wc1b, 256, 256, 256);
  cvt_pad_kernel<<<512,   256, 0, stream>>>(Wn1, Wn1b, 512, 512, 256);
  cvt_pad_kernel<<<256,   256, 0, stream>>>(Wn2, Wn2b, 256, 256, 256);

  egcl_edge_kernel<<<N_EDGES / 64, 256, 0, stream>>>(
      coord, eidx, hbf, We1p, be1, We2b, be2, Wc1b, bc1, Wc2,
      eij_out, agg, agg_c, cnt);

  egcl_node_kernel<<<(N_NODES + 63) / 64, 256, 0, stream>>>(
      h, coord, hbf, agg, Wn1b, bn1, Wn2b, bn2, ln_g, ln_b,
      agg_c, cnt, h_out, coord_out);
}
